// DeepSeekMLA_64518998720785
// MI455X (gfx1250) — matrix operand order repaired and checked by reading, compiled, not run on hardware
//
#include <hip/hip_runtime.h>
#include <hip/hip_bf16.h>

// ---------------------------------------------------------------------------
// DeepSeek MLA with weight absorption, register-blocked bf16 WMMA (gfx1250).
//
// Pass 0: one-time fp32 -> bf16 conversion of all operands (+ transpose of
//         the absorbed k-part of W_kv_up so every GEMM B-operand is K-unit-
//         stride and fragment loads become global_load_b128 pairs).
// Pass 1..6: absorbed-MLA dataflow (12 GF total instead of 79 GF + 1 GB kv
//         intermediate); all GEMMs on a 128x128 block / 64x64-per-wave WMMA
//         kernel: 16 v_wmma per 16 b128 fragment loads.
// Attention kernel keeps c_sel in LDS in BOTH orientations so every WMMA
// operand fragment is a pair of ds_load_b128 (no scalar u16 gathers).
// ---------------------------------------------------------------------------

typedef __attribute__((ext_vector_type(16))) __bf16 v16bf;
typedef __attribute__((ext_vector_type(8)))  float  v8f;

#define B_DIM    2
#define NQ_DIM   1024
#define NKV_DIM  2048
#define KSEL     64
#define D_DIM    1024
#define H_DIM    16
#define HD_DIM   64
#define LAT_DIM  128
#define ATTN_SCALE 0.125f   // 1/sqrt(64)

// Load a 16-element bf16 WMMA fragment as two 16-byte vector loads.
// A-operand (ISA: k = half*8 + (e&7) + ((e&8)<<1)): runs at +0 and +16 -> step 16.
// B-operand: same k order as A (base + half*8, runs at +0 and +16) -> step 16.
__device__ __forceinline__ v16bf load_frag(const __bf16* p, int step) {
    union { uint4 u[2]; v16bf v; } t;
    t.u[0] = *(const uint4*)(p);
    t.u[1] = *(const uint4*)(p + step);
    return t.v;
}

// ---------------------------------------------------------------------------
// fp32 -> bf16 elementwise convert (8 elems / thread, vectorized both sides)
// ---------------------------------------------------------------------------
__global__ __launch_bounds__(256) void cvt_f32_bf16(
    const float* __restrict__ src, __bf16* __restrict__ dst, int n)
{
    int i = (blockIdx.x * 256 + threadIdx.x) * 8;
    if (i >= n) return;
    float4 a = *(const float4*)(src + i);
    float4 b = *(const float4*)(src + i + 4);
    union { uint4 u; __bf16 h[8]; } o;
    o.h[0] = (__bf16)a.x; o.h[1] = (__bf16)a.y;
    o.h[2] = (__bf16)a.z; o.h[3] = (__bf16)a.w;
    o.h[4] = (__bf16)b.x; o.h[5] = (__bf16)b.y;
    o.h[6] = (__bf16)b.z; o.h[7] = (__bf16)b.w;
    *(uint4*)(dst + i) = o.u;
}

// Transposed convert of the absorbed k-part of W_kv_up:
//   dst[l * D + r] = (bf16) src[r * LAT + l],  r in [0,D), l in [0,LAT)
__global__ __launch_bounds__(256) void cvt_t_wku(
    const float* __restrict__ src, __bf16* __restrict__ dst)
{
    int i = blockIdx.x * 256 + threadIdx.x;      // D*LAT threads
    int r = i >> 7, l = i & (LAT_DIM - 1);
    dst[(size_t)l * D_DIM + r] = (__bf16)src[(size_t)r * LAT_DIM + l];
}

// ---------------------------------------------------------------------------
// Register-blocked bf16 GEMM:  C[m,n] = sum_k A[m*lda+k] * B[n*ldb+k]
// grid (ceil(N/128), M/128, Z), block 128 (4 waves, each a 64x64 C tile,
// 16 fp32 16x16 accumulators, 16 v_wmma per K-step of 32).
// ---------------------------------------------------------------------------
__global__ __launch_bounds__(128) void wmma_gemm_bf16(
    const __bf16* __restrict__ A, const __bf16* __restrict__ Bm,
    void* __restrict__ Cv, int f32out,
    int M, int N, int Kd, int lda, int ldb, int ldc,
    long long aZ, long long bZ, long long cZ)
{
    const int lane = threadIdx.x & 31;
    const int wave = threadIdx.x >> 5;
    const int half = lane >> 4;
    const int mn   = lane & 15;

    const int rowW = blockIdx.y * 128 + (wave >> 1) * 64;
    const int colW = blockIdx.x * 128 + (wave & 1) * 64;
    if (rowW >= M || colW >= N) return;   // wave-uniform

    const long long z = blockIdx.z;
    A  += z * aZ;
    Bm += z * bZ;

    const __bf16* arow[4];
    const __bf16* brow[4];
#pragma unroll
    for (int i = 0; i < 4; ++i) {
        arow[i] = A  + (size_t)(rowW + i * 16 + mn) * lda + half * 8;
        brow[i] = Bm + (size_t)(colW + i * 16 + mn) * ldb + half * 8;
    }

    const v8f vzero = {0.f, 0.f, 0.f, 0.f, 0.f, 0.f, 0.f, 0.f};
    v8f acc[4][4];
#pragma unroll
    for (int i = 0; i < 4; ++i)
#pragma unroll
        for (int j = 0; j < 4; ++j) acc[i][j] = vzero;

    for (int kk = 0; kk < Kd; kk += 32) {
        v16bf af[4], bf[4];
#pragma unroll
        for (int i = 0; i < 4; ++i) af[i] = load_frag(arow[i] + kk, 16);
#pragma unroll
        for (int j = 0; j < 4; ++j) bf[j] = load_frag(brow[j] + kk, 16);
#pragma unroll
        for (int i = 0; i < 4; ++i)
#pragma unroll
            for (int j = 0; j < 4; ++j)
                acc[i][j] = __builtin_amdgcn_wmma_f32_16x16x32_bf16(
                    false, af[i], false, bf[j], (short)0, acc[i][j], false, false);
    }

    if (f32out) {
        float* C = (float*)Cv + z * cZ;
#pragma unroll
        for (int i = 0; i < 4; ++i)
#pragma unroll
            for (int j = 0; j < 4; ++j)
#pragma unroll
                for (int r = 0; r < 8; ++r) {
                    int m = rowW + i * 16 + r + 8 * half;
                    int n = colW + j * 16 + mn;
                    C[(size_t)m * ldc + n] = acc[i][j][r];
                }
    } else {
        __bf16* C = (__bf16*)Cv + z * cZ;
#pragma unroll
        for (int i = 0; i < 4; ++i)
#pragma unroll
            for (int j = 0; j < 4; ++j)
#pragma unroll
                for (int r = 0; r < 8; ++r) {
                    int m = rowW + i * 16 + r + 8 * half;
                    int n = colW + j * 16 + mn;
                    C[(size_t)m * ldc + n] = (__bf16)acc[i][j][r];
                }
    }
}

// ---------------------------------------------------------------------------
// Fused MLA attention in latent space (one workgroup of 4 waves per query).
//   scores[h,k] = SCALE * sum_l qlat[h,l] * c_sel[k,l]   (16x64x128)
//   attn = softmax_k(scores);  u[h,l] = sum_k attn[h,k] * c_sel[k,l]
// c_sel kept in LDS in both orientations so all WMMA fragments are
// ds_load_b128 pairs. c_kv is bf16 (1 MB -> L2-resident on the 192 MB L2).
// ---------------------------------------------------------------------------
__global__ __launch_bounds__(128) void mla_attention(
    const __bf16* __restrict__ qlat,   // (B*NQ, H, LAT) bf16
    const __bf16* __restrict__ ckv,    // (B, NKV, LAT)  bf16
    const int*    __restrict__ indices,// (B, NQ, KSEL)
    __bf16* __restrict__ u)            // (B*NQ, H, LAT) bf16
{
    __shared__ __attribute__((aligned(16))) __bf16 sC [KSEL][LAT_DIM];  // 16 KB  [key][lat]
    __shared__ __attribute__((aligned(16))) __bf16 sCT[LAT_DIM][KSEL];  // 16 KB  [lat][key]
    __shared__ __attribute__((aligned(16))) __bf16 sQ[H_DIM][LAT_DIM];  //  4 KB
    __shared__ __attribute__((aligned(16))) float  sS[H_DIM][KSEL];     //  4 KB
    __shared__ __attribute__((aligned(16))) __bf16 sP[H_DIM][KSEL];     //  2 KB
    __shared__ int sIdx[KSEL];

    const int bq   = blockIdx.x;
    const int b    = bq / NQ_DIM;
    const int tid  = threadIdx.x;
    const int lane = tid & 31;
    const int wave = tid >> 5;
    const int half = lane >> 4;
    const int mn   = lane & 15;

    // stage indices once
    if (tid < KSEL) sIdx[tid] = indices[(size_t)bq * KSEL + tid];
    __syncthreads();

    // gather c_sel (64 x 128 bf16) via 16-byte copies; write both layouts
    for (int g = tid; g < KSEL * (LAT_DIM / 8); g += 128) {
        int key = g >> 4, l0 = (g & 15) * 8;
        int row = sIdx[key];
        union { uint4 u; __bf16 h[8]; } t;
        t.u = *(const uint4*)&ckv[((size_t)b * NKV_DIM + row) * LAT_DIM + l0];
        *(uint4*)&sC[key][l0] = t.u;
#pragma unroll
        for (int j = 0; j < 8; ++j) sCT[l0 + j][key] = t.h[j];
    }
    // q_lat tile (16 x 128 bf16)
    const __bf16* qrow = qlat + (size_t)bq * (H_DIM * LAT_DIM);
    for (int g = tid; g < (H_DIM * LAT_DIM) / 8; g += 128) {
        *(uint4*)&sQ[0][g * 8] = *(const uint4*)&qrow[g * 8];
    }
    __syncthreads();

    // scores: wave w -> key tile [16w, 16w+16), K over LAT (4 wmma)
    {
        v8f s = {0.f, 0.f, 0.f, 0.f, 0.f, 0.f, 0.f, 0.f};
        for (int kk = 0; kk < LAT_DIM; kk += 32) {
            v16bf a  = load_frag(&sQ[mn][kk + half * 8], 16);
            v16bf bb = load_frag(&sC[wave * 16 + mn][kk + half * 8], 16);
            s = __builtin_amdgcn_wmma_f32_16x16x32_bf16(
                false, a, false, bb, (short)0, s, false, false);
        }
#pragma unroll
        for (int r = 0; r < 8; ++r)
            sS[r + 8 * half][wave * 16 + mn] = s[r] * ATTN_SCALE;
    }
    __syncthreads();

    // softmax over 64 keys, one head per thread (threads 0..15)
    if (tid < H_DIM) {
        float mx = -3.4e38f;
        for (int k = 0; k < KSEL; ++k) mx = fmaxf(mx, sS[tid][k]);
        float sum = 0.f;
        for (int k = 0; k < KSEL; ++k) {
            float e = __expf(sS[tid][k] - mx);
            sS[tid][k] = e;
            sum += e;
        }
        float inv = 1.f / sum;
        for (int k = 0; k < KSEL; ++k)
            sP[tid][k] = (__bf16)(sS[tid][k] * inv);
    }
    __syncthreads();

    // u = attn @ c_sel : wave w covers latent cols [32w, 32w+32)
    __bf16* urow = u + (size_t)bq * (H_DIM * LAT_DIM);
#pragma unroll
    for (int t = 0; t < 2; ++t) {
        const int col0 = wave * 32 + t * 16;
        v8f acc = {0.f, 0.f, 0.f, 0.f, 0.f, 0.f, 0.f, 0.f};
        for (int kk = 0; kk < KSEL; kk += 32) {
            v16bf a  = load_frag(&sP[mn][kk + half * 8], 16);
            v16bf bb = load_frag(&sCT[col0 + mn][kk + half * 8], 16);
            acc = __builtin_amdgcn_wmma_f32_16x16x32_bf16(
                false, a, false, bb, (short)0, acc, false, false);
        }
#pragma unroll
        for (int r = 0; r < 8; ++r) {
            int h = r + 8 * half;
            urow[(size_t)h * LAT_DIM + col0 + mn] = (__bf16)acc[r];
        }
    }
}

// ---------------------------------------------------------------------------
extern "C" void kernel_launch(void* const* d_in, const int* in_sizes, int n_in,
                              void* d_out, int out_size, void* d_ws, size_t ws_size,
                              hipStream_t stream) {
    const float* x_q   = (const float*)d_in[0]; // (B, NQ, D)
    const float* x_kv  = (const float*)d_in[1]; // (B, NKV, D)
    const int*   idx   = (const int*)  d_in[2]; // (B, NQ, K)
    const float* W_q   = (const float*)d_in[3]; // (D, D)
    const float* W_kd  = (const float*)d_in[4]; // (LAT, D)
    const float* W_ku  = (const float*)d_in[5]; // (2D, LAT)
    const float* W_o   = (const float*)d_in[6]; // (D, D)
    float* out = (float*)d_out;                 // (B, NQ, D) fp32

    const int BQ = B_DIM * NQ_DIM;              // 2048
    // ---- bf16 workspace layout (all sizes multiples of 8 elements) --------
    __bf16* p = (__bf16*)d_ws;
    __bf16* xq_bf   = p; p += (size_t)BQ * D_DIM;                 // 2 M
    __bf16* xkv_bf  = p; p += (size_t)B_DIM * NKV_DIM * D_DIM;    // 4 M
    __bf16* Wq_bf   = p; p += (size_t)D_DIM * D_DIM;              // 1 M
    __bf16* Wkd_bf  = p; p += (size_t)LAT_DIM * D_DIM;            // 128 K
    __bf16* WkuT_bf = p; p += (size_t)LAT_DIM * D_DIM;            // 128 K (k-part, transposed)
    __bf16* Wv_bf   = p; p += (size_t)D_DIM * LAT_DIM;            // 128 K (v-part)
    __bf16* Wo_bf   = p; p += (size_t)D_DIM * D_DIM;              // 1 M
    __bf16* q_bf    = p; p += (size_t)BQ * D_DIM;                 // 2 M
    __bf16* qlat_bf = p; p += (size_t)BQ * H_DIM * LAT_DIM;       // 4 M
    __bf16* ckv_bf  = p; p += (size_t)B_DIM * NKV_DIM * LAT_DIM;  // 512 K
    __bf16* u_bf    = p; p += (size_t)BQ * H_DIM * LAT_DIM;       // 4 M
    __bf16* opre_bf = p; p += (size_t)BQ * D_DIM;                 // 2 M

    dim3 b256(256), b128(128);

    // ---- pass 0: convert everything to bf16 once ---------------------------
    cvt_f32_bf16<<<dim3(BQ * D_DIM / 2048), b256, 0, stream>>>(x_q,  xq_bf,  BQ * D_DIM);
    cvt_f32_bf16<<<dim3(B_DIM * NKV_DIM * D_DIM / 2048), b256, 0, stream>>>(x_kv, xkv_bf, B_DIM * NKV_DIM * D_DIM);
    cvt_f32_bf16<<<dim3(D_DIM * D_DIM / 2048), b256, 0, stream>>>(W_q,  Wq_bf,  D_DIM * D_DIM);
    cvt_f32_bf16<<<dim3(LAT_DIM * D_DIM / 2048), b256, 0, stream>>>(W_kd, Wkd_bf, LAT_DIM * D_DIM);
    cvt_f32_bf16<<<dim3(D_DIM * LAT_DIM / 2048), b256, 0, stream>>>(W_ku + (size_t)D_DIM * LAT_DIM, Wv_bf, D_DIM * LAT_DIM);
    cvt_t_wku   <<<dim3(D_DIM * LAT_DIM / 256), b256, 0, stream>>>(W_ku, WkuT_bf);
    cvt_f32_bf16<<<dim3(D_DIM * D_DIM / 2048), b256, 0, stream>>>(W_o,  Wo_bf,  D_DIM * D_DIM);

    // 1) q = x_q @ W_q.T : M=2048 N=1024 K=1024 -> bf16
    wmma_gemm_bf16<<<dim3(D_DIM / 128, BQ / 128, 1), b128, 0, stream>>>(
        xq_bf, Wq_bf, q_bf, 0, BQ, D_DIM, D_DIM, D_DIM, D_DIM, D_DIM,
        0LL, 0LL, 0LL);

    // 2) q_lat[., h, :] = q[., h*64:(h+1)*64] @ W_uk_h  (z = head)
    //    B[n=l][k=d] = WkuT[l*D + h*64 + d]  -> ldb = D, bZ = 64
    wmma_gemm_bf16<<<dim3(1, BQ / 128, H_DIM), b128, 0, stream>>>(
        q_bf, WkuT_bf, qlat_bf, 0, BQ, LAT_DIM, HD_DIM,
        D_DIM, D_DIM, H_DIM * LAT_DIM,
        (long long)HD_DIM, (long long)HD_DIM, (long long)LAT_DIM);

    // 3) c_kv = x_kv @ W_kv_down.T : M=4096 N=128 K=1024 -> bf16
    wmma_gemm_bf16<<<dim3(1, (B_DIM * NKV_DIM) / 128, 1), b128, 0, stream>>>(
        xkv_bf, Wkd_bf, ckv_bf, 0, B_DIM * NKV_DIM, LAT_DIM, D_DIM,
        D_DIM, D_DIM, LAT_DIM, 0LL, 0LL, 0LL);

    // 4) fused gather + scores + softmax + latent-weighted sum
    mla_attention<<<dim3(BQ), b128, 0, stream>>>(qlat_bf, ckv_bf, idx, u_bf);

    // 5) out_pre[., h*64:] = u[., h, :] @ W_v_h.T  (z = head)
    //    B[n=d][k=l] = Wv_bf[(h*64+d)*128 + l] -> ldb = 128, bZ = 64*128
    wmma_gemm_bf16<<<dim3(1, BQ / 128, H_DIM), b128, 0, stream>>>(
        u_bf, Wv_bf, opre_bf, 0, BQ, HD_DIM, LAT_DIM,
        H_DIM * LAT_DIM, LAT_DIM, D_DIM,
        (long long)LAT_DIM, (long long)HD_DIM * LAT_DIM, (long long)HD_DIM);

    // 6) out = out_pre @ W_out.T : M=2048 N=1024 K=1024 -> fp32 d_out
    wmma_gemm_bf16<<<dim3(D_DIM / 128, BQ / 128, 1), b128, 0, stream>>>(
        opre_bf, Wo_bf, out, 1, BQ, D_DIM, D_DIM, D_DIM, D_DIM, D_DIM,
        0LL, 0LL, 0LL);
}
